// Encoder_28999619182810
// MI455X (gfx1250) — compile-verified, matrix operand order checked by reading, not run on hardware
//
#include <hip/hip_runtime.h>
#include <hip/hip_bf16.h>

// Problem sizes
#define B_  64
#define S_  512
#define E_  1024
#define H_  1024
#define G4  4096           // 4*H
#define T_  (B_*S_)        // 32768 tokens

typedef __attribute__((ext_vector_type(16))) __bf16 v16bf;
typedef __attribute__((ext_vector_type(8)))  float  v8f;
typedef __attribute__((ext_vector_type(4)))  float  f32x4;   // 16-byte chunk
typedef __attribute__((ext_vector_type(4)))  __bf16 bf16x4;
typedef __attribute__((ext_vector_type(4)))  unsigned int u32x4;
typedef __attribute__((ext_vector_type(8)))  int i32x8;
typedef __attribute__((ext_vector_type(4)))  int i32x4;

#if defined(__has_builtin)
#if __has_builtin(__builtin_amdgcn_tensor_load_to_lds) && \
    __has_builtin(__builtin_amdgcn_s_wait_tensorcnt)
#define USE_TDM 1
#endif
#endif
#ifndef USE_TDM
#define USE_TDM 0
#endif

union Frag { v16bf v; f32x4 f[2]; };

__device__ __forceinline__ v8f wmma_bf16(v16bf a, v16bf b, v8f c) {
    // 8 args: (neg_a, A, neg_b, B, c_mod, C, reuse_a, reuse_b)
    return __builtin_amdgcn_wmma_f32_16x16x32_bf16(false, a, false, b, (short)0, c,
                                                   false, false);
}

__device__ __forceinline__ float sig(float x) { return 1.f / (1.f + __expf(-x)); }

// ---------------------------------------------------------------------------
// Kernel 1: W_ih, W_hh [K=1024][N=4096] f32 -> transposed bf16 [N][K]
// ---------------------------------------------------------------------------
__global__ __launch_bounds__(256) void prep_wt(const float* __restrict__ Wih,
                                               const float* __restrict__ Whh,
                                               __bf16* __restrict__ WihT,
                                               __bf16* __restrict__ WhhT) {
    const int total = E_ * G4;
    for (int i = blockIdx.x * blockDim.x + threadIdx.x; i < total;
         i += gridDim.x * blockDim.x) {
        const int n = i >> 10;        // output row (gate column), 0..4095
        const int k = i & 1023;       // output col (contraction), 0..1023
        WihT[i] = (__bf16)Wih[(size_t)k * G4 + n];
        WhhT[i] = (__bf16)Whh[(size_t)k * G4 + n];
    }
}

// ---------------------------------------------------------------------------
// Kernel 2: embedding gather + f32->bf16 cast, time-major rows t = s*B + b
// ---------------------------------------------------------------------------
__global__ __launch_bounds__(256) void embed_cast(const int* __restrict__ src,
                                                  const float* __restrict__ emb,
                                                  __bf16* __restrict__ X) {
    const int t = blockIdx.x;             // 0..32767
    const int s = t >> 6, b = t & 63;
    const int row = src[b * S_ + s];
    const f32x4 v =
        reinterpret_cast<const f32x4*>(emb + (size_t)row * E_)[threadIdx.x];
    bf16x4 o;
    o[0] = (__bf16)v[0]; o[1] = (__bf16)v[1];
    o[2] = (__bf16)v[2]; o[3] = (__bf16)v[3];
    *reinterpret_cast<bf16x4*>(X + (size_t)t * E_ + threadIdx.x * 4) = o;
}

// ---------------------------------------------------------------------------
// Kernel 3: GEMM1  Xg[t][n] = X[t][:] . WihT[n][:] + bias[n]   (f32 out)
// M=32768, N=4096, K=1024.  Workgroup tile 128x128, 8 waves (4M x 2N),
// wave tile 32x64 = 2x4 sub-tiles of 16x16.
// Tile staging via Tensor Data Mover, DOUBLE-BUFFERED: while all 8 waves run
// WMMAs on LDS buffer i&1, the TDM DMAs tile i+1 into buffer (i+1)&1;
// s_wait_tensorcnt<=1 (in-order completion) guarantees tile i has landed.
// D# pad fields insert 32B padding per 64B row -> LDS row stride 96B (48 bf16)
// so WMMA fragments remain two aligned ds_load_b128 per lane.
// ---------------------------------------------------------------------------
__global__ __launch_bounds__(256) void gemm_xw(const __bf16* __restrict__ X,
                                               const __bf16* __restrict__ WT,
                                               const float* __restrict__ bias,
                                               float* __restrict__ Xg) {
    __shared__ alignas(16) __bf16 As[2][128 * 48];   // 2 x 12KB
    __shared__ alignas(16) __bf16 Bs[2][128 * 48];

    const int tid  = threadIdx.x;
    const int lane = tid & 31, wave = tid >> 5;
    const int wrow = wave >> 1, wcol = wave & 1;
    const int half = lane >> 4, l15 = lane & 15;
    const int Mbase = blockIdx.y * 128;
    const int Nbase = blockIdx.x * 128;
    const int KSTEPS = E_ / 32;

    v8f acc[2][4];
#pragma unroll
    for (int i = 0; i < 2; ++i)
#pragma unroll
        for (int j = 0; j < 4; ++j) acc[i][j] = (v8f){0.f,0.f,0.f,0.f,0.f,0.f,0.f,0.f};

#if USE_TDM
    // --- Tensor DMA descriptors (CDNA5 ISA D# layout) ---------
    const unsigned ldsA = (unsigned)(size_t)&As[0][0];
    const unsigned ldsB = (unsigned)(size_t)&Bs[0][0];
    const unsigned bufBytes = 128u * 48u * 2u;       // 12KB per buffer
    const unsigned long long gA0 =
        (unsigned long long)(size_t)(X + (size_t)Mbase * E_);
    const unsigned long long gB0 =
        (unsigned long long)(size_t)(WT + (size_t)Nbase * E_);
    // Group1: data_size=2B (1), pad_enable, pad_interval=3 (16 dwords),
    // pad_amount=7 (8 dwords); tensor 32x128, tile 32x128, dim0 stride 1024.
    i32x8 g1;
    g1[0] = (1 << 16) | (1 << 20) | (3 << 22) | (7 << 25);
    g1[1] = (int)(32u << 16);          // tensor_dim0 = 32 (bits 79:48 low half)
    g1[2] = (int)(128u << 16);         // tensor_dim1 = 128 (bits 111:80 low half)
    g1[3] = (int)(32u << 16);          // tile_dim0 = 32 (bits 127:112)
    g1[4] = 128;                       // tile_dim1 = 128, tile_dim2 = 0
    g1[5] = 1024;                      // tensor_dim0_stride = 1024 elements
    g1[6] = 0;
    g1[7] = 0;
    const i32x4 gz4 = {0, 0, 0, 0};            // groups 2/3 unused (2-D tile)
    const i32x8 gz8 = {0, 0, 0, 0, 0, 0, 0, 0};

    const auto issue_tiles = [&](unsigned buf, int kk) {
        const unsigned long long ga = gA0 + (unsigned long long)kk * 2u;
        u32x4 d0a;
        d0a[0] = 1u;                                   // count=1
        d0a[1] = ldsA + buf * bufBytes;                // lds_addr
        d0a[2] = (unsigned)ga;                         // global_addr[31:0]
        d0a[3] = (unsigned)((ga >> 32) & 0x1FFFFFFull) // global_addr[56:32]
                 | (2u << 30);                         // type=2 ("image")
        __builtin_amdgcn_tensor_load_to_lds(d0a, g1, gz4, gz4, gz8, 0);

        const unsigned long long gb = gB0 + (unsigned long long)kk * 2u;
        u32x4 d0b;
        d0b[0] = 1u;
        d0b[1] = ldsB + buf * bufBytes;
        d0b[2] = (unsigned)gb;
        d0b[3] = (unsigned)((gb >> 32) & 0x1FFFFFFull) | (2u << 30);
        __builtin_amdgcn_tensor_load_to_lds(d0b, g1, gz4, gz4, gz8, 0);
    };

    // Prologue: start DMA of tile 0 (scalar branch: TDM ignores EXEC, so only
    // wave 0 may reach the instruction).
    if (__builtin_amdgcn_readfirstlane((int)(threadIdx.x >> 5)) == 0) {
        issue_tiles(0u, 0);
    }
#else
    const int rs = tid >> 1;              // staging row 0..127
    const int kc = (tid & 1) * 16;        // staging k-chunk 0/16
#endif

    for (int i = 0; i < KSTEPS; ++i) {
        const int buf = i & 1;
        __syncthreads();   // all waves done with buffer (i+1)&1 from iter i-1
#if USE_TDM
        if (__builtin_amdgcn_readfirstlane((int)(threadIdx.x >> 5)) == 0) {
            if (i + 1 < KSTEPS) {
                issue_tiles((unsigned)((i + 1) & 1), (i + 1) * 32);
                __builtin_amdgcn_s_wait_tensorcnt(1);  // tile i landed (in-order)
            } else {
                __builtin_amdgcn_s_wait_tensorcnt(0);  // last tile
            }
        }
#else
        {
            const int kk = i * 32;
            const f32x4* g = reinterpret_cast<const f32x4*>(
                X + (size_t)(Mbase + rs) * E_ + kk + kc);
            *reinterpret_cast<f32x4*>(&As[buf][rs * 48 + kc])     = g[0];
            *reinterpret_cast<f32x4*>(&As[buf][rs * 48 + kc + 8]) = g[1];
            const f32x4* h = reinterpret_cast<const f32x4*>(
                WT + (size_t)(Nbase + rs) * E_ + kk + kc);
            *reinterpret_cast<f32x4*>(&Bs[buf][rs * 48 + kc])     = h[0];
            *reinterpret_cast<f32x4*>(&Bs[buf][rs * 48 + kc + 8]) = h[1];
        }
#endif
        __syncthreads();   // tile i visible in LDS

        const __bf16* Ab = &As[buf][0];
        const __bf16* Bb = &Bs[buf][0];
        Frag a[2], b[4];
#pragma unroll
        for (int it = 0; it < 2; ++it) {
            const int m = wrow * 32 + it * 16 + l15;
            a[it].f[0] = *reinterpret_cast<const f32x4*>(&Ab[m * 48 + half * 8]);
            a[it].f[1] = *reinterpret_cast<const f32x4*>(&Ab[m * 48 + 16 + half * 8]);
        }
#pragma unroll
        for (int jt = 0; jt < 4; ++jt) {
            const int n = wcol * 64 + jt * 16 + l15;
            b[jt].f[0] = *reinterpret_cast<const f32x4*>(&Bb[n * 48 + half * 8]);
            b[jt].f[1] = *reinterpret_cast<const f32x4*>(&Bb[n * 48 + 16 + half * 8]);
        }
#pragma unroll
        for (int it = 0; it < 2; ++it)
#pragma unroll
            for (int jt = 0; jt < 4; ++jt)
                acc[it][jt] = wmma_bf16(a[it].v, b[jt].v, acc[it][jt]);
    }

    // Epilogue: += bias, store f32 (C/D layout: elem r -> M = r + 8*half, N = l15)
#pragma unroll
    for (int it = 0; it < 2; ++it)
#pragma unroll
        for (int jt = 0; jt < 4; ++jt) {
            const int ng = Nbase + wcol * 64 + jt * 16 + l15;
            const float bv = bias[ng];
#pragma unroll
            for (int r = 0; r < 8; ++r) {
                const int m = Mbase + wrow * 32 + it * 16 + r + 8 * half;
                Xg[(size_t)m * G4 + ng] = acc[it][jt][r] + bv;
            }
        }
}

// ---------------------------------------------------------------------------
// Kernel 4: zero h ping-pong buffers + grid-barrier counter (every call, so
// graph replays are deterministic).
// ---------------------------------------------------------------------------
__global__ __launch_bounds__(256) void init_state(__bf16* __restrict__ hbuf,
                                                  unsigned* __restrict__ cnt) {
    const int i = blockIdx.x * blockDim.x + threadIdx.x;   // 65536 dwords
    reinterpret_cast<unsigned*>(hbuf)[i] = 0u;             // 2 * 64*1024 bf16
    if (i == 0) *cnt = 0u;
}

// ---------------------------------------------------------------------------
// Kernel 5: persistent LSTM recurrence.  64 workgroups x 256 threads.
// Workgroup owns hidden columns [hbase, hbase+16) across all 4 gates.
// Per step: GEMM tile M=64 x N=64 (4 m-tiles x 4 gate-tiles, 2 tiles/wave),
// K=1024 from L2-resident WhhT + double-buffered bf16 h.  Gates combined via
// LDS; c stays in registers for the whole sequence; one grid barrier/step.
// ---------------------------------------------------------------------------
__global__ __launch_bounds__(256) void lstm_persist(const __bf16* __restrict__ WhhT,
                                                    const float* __restrict__ Xg,
                                                    __bf16* __restrict__ hbuf,
                                                    float* __restrict__ out,
                                                    unsigned* __restrict__ cnt) {
    __shared__ float gld[4][B_][16];     // [gate][batch][local h] = 16 KB

    const int tid  = threadIdx.x;
    const int lane = tid & 31, wave = tid >> 5;
    const int mtile = wave & 3;          // batch rows mtile*16..+15
    const int gp    = (wave >> 2) * 2;   // this wave handles gates gp, gp+1
    const int half  = lane >> 4, l15 = lane & 15;
    const int hbase = blockIdx.x * 16;

    float creg[4] = {0.f, 0.f, 0.f, 0.f};   // cell state lives in registers

    const __bf16* Wb0 = WhhT + (size_t)((gp)     * H_ + hbase + l15) * H_ + half * 8;
    const __bf16* Wb1 = WhhT + (size_t)((gp + 1) * H_ + hbase + l15) * H_ + half * 8;

    for (int s = 0; s < S_; ++s) {
        const __bf16* hcur = hbuf + (size_t)(s & 1) * (B_ * H_);
        __bf16*       hnxt = hbuf + (size_t)((s + 1) & 1) * (B_ * H_);

        v8f acc0 = (v8f){0.f,0.f,0.f,0.f,0.f,0.f,0.f,0.f};
        v8f acc1 = (v8f){0.f,0.f,0.f,0.f,0.f,0.f,0.f,0.f};

        const __bf16* ha = hcur + (size_t)(mtile * 16 + l15) * H_ + half * 8;
#pragma unroll 4
        for (int kk = 0; kk < H_; kk += 32) {
            Frag a, b0, b1;
            a.f[0]  = *reinterpret_cast<const f32x4*>(ha  + kk);
            a.f[1]  = *reinterpret_cast<const f32x4*>(ha  + kk + 16);
            b0.f[0] = *reinterpret_cast<const f32x4*>(Wb0 + kk);
            b0.f[1] = *reinterpret_cast<const f32x4*>(Wb0 + kk + 16);
            b1.f[0] = *reinterpret_cast<const f32x4*>(Wb1 + kk);
            b1.f[1] = *reinterpret_cast<const f32x4*>(Wb1 + kk + 16);
            acc0 = wmma_bf16(a.v, b0.v, acc0);
            acc1 = wmma_bf16(a.v, b1.v, acc1);
        }

        // Add precomputed input preactivations, stage into LDS gate buffer.
        const float* xrow = Xg + (size_t)s * B_ * G4;
        const int n0 = (gp)     * H_ + hbase + l15;
        const int n1 = (gp + 1) * H_ + hbase + l15;
#pragma unroll
        for (int r = 0; r < 8; ++r) {
            const int m = mtile * 16 + r + 8 * half;
            gld[gp][m][l15]     = acc0[r] + xrow[(size_t)m * G4 + n0];
            gld[gp + 1][m][l15] = acc1[r] + xrow[(size_t)m * G4 + n1];
        }
        __syncthreads();

        // Elementwise gate combine: 1024 cells, 4 per thread (fixed ownership
        // across steps so creg is valid).
#pragma unroll
        for (int j = 0; j < 4; ++j) {
            const int cell = tid + 256 * j;
            const int b = cell >> 4, hh = cell & 15;
            const float gi = gld[0][b][hh];
            const float gf = gld[1][b][hh];
            const float gg = gld[2][b][hh];
            const float go = gld[3][b][hh];
            const float cn = sig(gf) * creg[j] + sig(gi) * tanhf(gg);
            creg[j] = cn;
            const float hn = sig(go) * tanhf(cn);
            out[(size_t)b * (S_ * H_) + (size_t)s * H_ + hbase + hh] = hn;
            hnxt[(size_t)b * H_ + hbase + hh] = (__bf16)hn;
        }

        // Device-wide barrier: publish h(t+1), wait for all 64 workgroups.
        __threadfence();
        __syncthreads();
        if (tid == 0) {
            atomicAdd(cnt, 1u);
            const unsigned target = (unsigned)(s + 1) * 64u;
            while (*(volatile unsigned*)cnt < target) __builtin_amdgcn_s_sleep(2);
        }
        __syncthreads();
    }
}

// ---------------------------------------------------------------------------
// Launch.  Workspace layout (~593 MB):
//   WihT 8MB | WhhT 8MB | X 64MB | Xg 512MB | hbuf 256KB | cnt
// ---------------------------------------------------------------------------
extern "C" void kernel_launch(void* const* d_in, const int* in_sizes, int n_in,
                              void* d_out, int out_size, void* d_ws, size_t ws_size,
                              hipStream_t stream) {
    const int*   src  = (const int*)  d_in[0];
    const float* emb  = (const float*)d_in[1];
    const float* Wih  = (const float*)d_in[2];
    const float* Whh  = (const float*)d_in[3];
    const float* bias = (const float*)d_in[4];
    float* out = (float*)d_out;

    char* ws = (char*)d_ws;
    size_t off = 0;
    __bf16* WihT = (__bf16*)(ws + off); off += (size_t)E_ * G4 * 2;
    __bf16* WhhT = (__bf16*)(ws + off); off += (size_t)E_ * G4 * 2;
    __bf16* X    = (__bf16*)(ws + off); off += (size_t)T_ * E_ * 2;
    float*  Xg   = (float*) (ws + off); off += (size_t)T_ * G4 * 4;
    __bf16* hbuf = (__bf16*)(ws + off); off += (size_t)2 * B_ * H_ * 2;
    unsigned* cnt = (unsigned*)(ws + off); off += 256;
    (void)ws_size; (void)in_sizes; (void)n_in; (void)out_size;

    prep_wt   <<<4096, 256, 0, stream>>>(Wih, Whh, WihT, WhhT);
    embed_cast<<<T_,   256, 0, stream>>>(src, emb, X);
    dim3 g1(G4 / 128, T_ / 128);           // 32 x 256 workgroups
    gemm_xw   <<<g1,   256, 0, stream>>>(X, WihT, bias, Xg);
    init_state<<<256,  256, 0, stream>>>(hbuf, cnt);
    lstm_persist<<<64, 256, 0, stream>>>(WhhT, Xg, hbuf, out, cnt);
}
